// Perceiver_44306882625602
// MI455X (gfx1250) — matrix operand order repaired and checked by reading, not compiled, not run on hardware
//
#include <hip/hip_runtime.h>
#include <hip/hip_bf16.h>

typedef __bf16 bf16_t;
typedef __attribute__((ext_vector_type(16))) __bf16 v16bf;
typedef __attribute__((ext_vector_type(8)))  __bf16 v8bf;
typedef __attribute__((ext_vector_type(8)))  float  v8f;

// ---------------------------------------------------------------------------
// Generic batched GEMM:  C[b] = A[b] (MxK, bf16) * B[b] (KxN, bf16) + bias
//   A row-major (lda).  B row-major (ldb) or, if TRANSB, B is N x K row-major
//   so that logical B[k][n] = Bp[n*ldb + k]  (used for Q @ K^T).
//   C written as f32 or bf16 (OUTBF).
//   Two batch dims (nb1 x nb2) with independent strides for A/B/C.
//
// REQUIRES (guaranteed by every launch in this file):
//   M % 128 == 0, N % 128 == 0, K % 32 == 0, lda/ldb/ldc % 8 == 0,
//   all base pointers + batch strides 16B aligned.
//
// Block tile 128x128, 256 threads = 8 wave32 (2 along M x 4 along N),
// each wave computes 64x32 via 4x2 WMMA accumulators.
// ---------------------------------------------------------------------------
#define BM 128
#define BN 128
#define BK 32
#define LDS_STRIDE 40  // 32 + 8 pad; 40*2B = 80B keeps 16B alignment per row

template <bool TRANSB, bool OUTBF>
__global__ __launch_bounds__(256) void k_gemm(
    const bf16_t* __restrict__ A, const bf16_t* __restrict__ B,
    void* __restrict__ C, const float* __restrict__ bias,
    int M, int N, int K, int lda, int ldb, int ldc,
    long long sA1, long long sB1, long long sC1,
    long long sA2, long long sB2, long long sC2, int nb2) {
  __shared__ __align__(16) bf16_t As[BM * LDS_STRIDE];
  __shared__ __align__(16) bf16_t Bs[BN * LDS_STRIDE];

  const int b1 = blockIdx.z / nb2;
  const int b2 = blockIdx.z % nb2;
  A += (long long)b1 * sA1 + (long long)b2 * sA2;
  B += (long long)b1 * sB1 + (long long)b2 * sB2;
  const long long coff = (long long)b1 * sC1 + (long long)b2 * sC2;

  const int m0 = blockIdx.y * BM;
  const int n0 = blockIdx.x * BN;
  const int tid = threadIdx.x;
  const int wave = tid >> 5;
  const int lane = tid & 31;
  const int wm = (wave & 1) * 64;   // 2 waves along M, 64 rows each
  const int wn = (wave >> 1) * 32;  // 4 waves along N, 32 cols each
  const int lm = lane & 15;
  const int lh = lane >> 4;

  // Per-thread staging coordinates (2 chunks of 8 bf16 per tile per thread).
  const int a_row0 = tid >> 2;            // chunk0 row,  +64 for chunk1
  const int a_col = (tid & 3) * 8;        // k offset within tile
  const int bt_n0 = tid >> 2;             // TRANSB: n row in tile
  const int bt_col = (tid & 3) * 8;       // TRANSB: k offset
  const int bn_k0 = tid >> 4;             // !TRANSB: k row (0..15, +16)
  const int bn_n0 = (tid & 15) * 8;       // !TRANSB: n offset

  v8f acc[4][2] = {};

  for (int k0 = 0; k0 < K; k0 += BK) {
    // ---- prefetch tiles two k-steps ahead (global_prefetch_b8) ----
    if (k0 + 2 * BK < K) {
      __builtin_prefetch(&A[(long long)(m0 + a_row0) * lda + (k0 + 2 * BK) + a_col], 0, 1);
      if (TRANSB)
        __builtin_prefetch(&B[(long long)(n0 + bt_n0) * ldb + (k0 + 2 * BK) + bt_col], 0, 1);
      else
        __builtin_prefetch(&B[(long long)(k0 + 2 * BK + bn_k0) * ldb + (n0 + bn_n0)], 0, 1);
    }

    // ---- stage A tile (BM x BK): 2 x 16B vector copies per thread ----
    #pragma unroll
    for (int t = 0; t < 2; ++t) {
      int row = a_row0 + t * 64;
      v8bf v = *(const v8bf*)&A[(long long)(m0 + row) * lda + (k0 + a_col)];
      *(v8bf*)&As[row * LDS_STRIDE + a_col] = v;
    }
    // ---- stage B tile transposed into LDS: Bs[n][k] ----
    if (TRANSB) {
      // memory already N-major with contiguous k: straight vector copies
      #pragma unroll
      for (int t = 0; t < 2; ++t) {
        int nn = bt_n0 + t * 64;
        v8bf v = *(const v8bf*)&B[(long long)(n0 + nn) * ldb + (k0 + bt_col)];
        *(v8bf*)&Bs[nn * LDS_STRIDE + bt_col] = v;
      }
    } else {
      // memory is K-major: vector load along n, scalar scatter into Bs[n][k]
      #pragma unroll
      for (int t = 0; t < 2; ++t) {
        int kk = bn_k0 + t * 16;
        v8bf v = *(const v8bf*)&B[(long long)(k0 + kk) * ldb + (n0 + bn_n0)];
        #pragma unroll
        for (int j = 0; j < 8; ++j) Bs[(bn_n0 + j) * LDS_STRIDE + kk] = v[j];
      }
    }
    __syncthreads();

    // ---- build fragments (documented gfx1250 wave32 layouts) ----
    // A: lane holds row (lane&15); elems 0..7 = K half*8.., 8..15 = K 16+half*8..
    v16bf a_frag[4];
    #pragma unroll
    for (int tm = 0; tm < 4; ++tm) {
      const bf16_t* ap = &As[(wm + tm * 16 + lm) * LDS_STRIDE];
      v8bf lo = *(const v8bf*)&ap[lh * 8];
      v8bf hi = *(const v8bf*)&ap[16 + lh * 8];
      a_frag[tm] = __builtin_shufflevector(lo, hi, 0, 1, 2, 3, 4, 5, 6, 7, 8,
                                           9, 10, 11, 12, 13, 14, 15);
    }
    // B: lane holds col (lane&15); elems 0..7 = K half*8.., 8..15 = K 16+half*8..
    v16bf b_frag[2];
    #pragma unroll
    for (int tn = 0; tn < 2; ++tn) {
      const bf16_t* bp = &Bs[(wn + tn * 16 + lm) * LDS_STRIDE];
      v8bf lo = *(const v8bf*)&bp[lh * 8];
      v8bf hi = *(const v8bf*)&bp[16 + lh * 8];
      b_frag[tn] = __builtin_shufflevector(lo, hi, 0, 1, 2, 3, 4, 5, 6, 7, 8,
                                           9, 10, 11, 12, 13, 14, 15);
    }

    #pragma unroll
    for (int tm = 0; tm < 4; ++tm)
      #pragma unroll
      for (int tn = 0; tn < 2; ++tn)
        acc[tm][tn] = __builtin_amdgcn_wmma_f32_16x16x32_bf16(
            false, a_frag[tm], false, b_frag[tn], (short)0, acc[tm][tn],
            false, false);
    __syncthreads();
  }

  // ---- epilogue: C/D layout — vgpr r holds row half*8 + r, column = lane&15
  #pragma unroll
  for (int tm = 0; tm < 4; ++tm) {
    #pragma unroll
    for (int tn = 0; tn < 2; ++tn) {
      int n = n0 + wn + tn * 16 + lm;
      float bv = bias ? bias[n] : 0.0f;
      #pragma unroll
      for (int r = 0; r < 8; ++r) {
        int m = m0 + wm + tm * 16 + lh * 8 + r;
        float val = acc[tm][tn][r] + bv;
        long long idx = coff + (long long)m * ldc + n;
        if (OUTBF) ((bf16_t*)C)[idx] = (bf16_t)val;
        else       ((float*)C)[idx]  = val;
      }
    }
  }
}

// ---------------------------------------------------------------------------
// Elementwise / reduction kernels
// ---------------------------------------------------------------------------
__global__ __launch_bounds__(256) void k_cvt_bf16(const float* __restrict__ s,
                                                  bf16_t* __restrict__ d,
                                                  long long n) {
  long long i = (long long)blockIdx.x * 256 + threadIdx.x;
  if (i < n) d[i] = (bf16_t)s[i];
}

// tokens[b,n,e] = sum_c x[b,c,n]*w[e,c] + conv_b[e] + pos[n,e]   (C=3)
__global__ __launch_bounds__(256) void k_conv_pos(const float* __restrict__ x,
                                                  const float* __restrict__ w,
                                                  const float* __restrict__ cb,
                                                  const float* __restrict__ pos,
                                                  bf16_t* __restrict__ tok) {
  long long i = (long long)blockIdx.x * 256 + threadIdx.x;  // B*N*E
  if (i >= (long long)4 * 4096 * 256) return;
  int e = (int)(i & 255);
  long long bn = i >> 8;
  int n = (int)(bn & 4095);
  int b = (int)(bn >> 12);
  const float* xp = x + ((long long)b * 3) * 4096 + n;
  float s = cb[e] + pos[(long long)n * 256 + e] + xp[0] * w[e * 3 + 0] +
            xp[4096] * w[e * 3 + 1] + xp[8192] * w[e * 3 + 2];
  tok[i] = (bf16_t)s;
}

// exact GELU (erf form), f32 in -> bf16 out
__global__ __launch_bounds__(256) void k_gelu_bf16(const float* __restrict__ s,
                                                   bf16_t* __restrict__ d,
                                                   long long n) {
  long long i = (long long)blockIdx.x * 256 + threadIdx.x;
  if (i < n) {
    float v = s[i];
    d[i] = (bf16_t)(0.5f * v * (1.0f + erff(v * 0.70710678118654752f)));
  }
}

// In-place row softmax over PER*256 bf16 values, with logit scale.
template <int PER>
__global__ __launch_bounds__(256) void k_softmax(bf16_t* __restrict__ s,
                                                 float scale) {
  long long row = blockIdx.x;
  bf16_t* p = s + row * (long long)(PER * 256);
  int tid = threadIdx.x;
  float v[PER];
  float mx = -3.0e38f;
  #pragma unroll
  for (int i = 0; i < PER; ++i) {
    v[i] = (float)p[tid + i * 256] * scale;
    mx = fmaxf(mx, v[i]);
  }
  __shared__ float red[256];
  red[tid] = mx;
  __syncthreads();
  for (int o = 128; o > 0; o >>= 1) {
    if (tid < o) red[tid] = fmaxf(red[tid], red[tid + o]);
    __syncthreads();
  }
  mx = red[0];
  __syncthreads();
  float sum = 0.0f;
  #pragma unroll
  for (int i = 0; i < PER; ++i) {
    v[i] = __expf(v[i] - mx);
    sum += v[i];
  }
  red[tid] = sum;
  __syncthreads();
  for (int o = 128; o > 0; o >>= 1) {
    if (tid < o) red[tid] += red[tid + o];
    __syncthreads();
  }
  float inv = 1.0f / red[0];
  #pragma unroll
  for (int i = 0; i < PER; ++i) p[tid + i * 256] = (bf16_t)(v[i] * inv);
}

// out = LayerNorm(x + res) over E=256; res row index optionally mod res_mod
// (for broadcast latents residual). Writes f32 and bf16 copies.
__global__ __launch_bounds__(256) void k_add_ln(const float* __restrict__ x,
                                                const float* __restrict__ res,
                                                int res_mod,
                                                const float* __restrict__ g,
                                                const float* __restrict__ bt,
                                                float* __restrict__ outf,
                                                bf16_t* __restrict__ outb) {
  int row = blockIdx.x;
  int e = threadIdx.x;
  long long xi = (long long)row * 256 + e;
  long long ri = (long long)(res_mod ? (row % res_mod) : row) * 256 + e;
  float v = x[xi] + res[ri];
  __shared__ float red[256];
  red[e] = v;
  __syncthreads();
  for (int o = 128; o > 0; o >>= 1) {
    if (e < o) red[e] += red[e + o];
    __syncthreads();
  }
  float mean = red[0] * (1.0f / 256.0f);
  __syncthreads();
  float d = v - mean;
  red[e] = d * d;
  __syncthreads();
  for (int o = 128; o > 0; o >>= 1) {
    if (e < o) red[e] += red[e + o];
    __syncthreads();
  }
  float var = red[0] * (1.0f / 256.0f);
  float y = g[e] * d * rsqrtf(var + 1e-5f) + bt[e];
  outf[xi] = y;
  outb[xi] = (bf16_t)y;
}

// zmean[b,e] = mean_q z[b*512+q, e]
__global__ __launch_bounds__(256) void k_mean(const float* __restrict__ z,
                                              float* __restrict__ zmean) {
  int b = blockIdx.x;
  int e = threadIdx.x;
  float s = 0.0f;
  for (int q = 0; q < 512; ++q) s += z[((long long)b * 512 + q) * 256 + e];
  zmean[b * 256 + e] = s * (1.0f / 512.0f);
}

// out[b,n] = zmean[b,:] @ cls_w[:,n] + cls_b[n]   (N=1000, tiny: scalar)
__global__ __launch_bounds__(256) void k_cls(const float* __restrict__ zmean,
                                             const float* __restrict__ w,
                                             const float* __restrict__ bias,
                                             float* __restrict__ out) {
  int n = blockIdx.x * 256 + threadIdx.x;
  int b = blockIdx.y;
  if (n >= 1000) return;
  const float* zm = zmean + b * 256;
  float s = bias[n];
  for (int e = 0; e < 256; ++e) s += zm[e] * w[(long long)e * 1000 + n];
  out[(long long)b * 1000 + n] = s;
}

// ---------------------------------------------------------------------------
// Host-side launch helpers
// ---------------------------------------------------------------------------
static void gemm(hipStream_t st, bool transb, bool outbf, const bf16_t* A,
                 const bf16_t* B, void* C, const float* bias, int M, int N,
                 int K, int lda, int ldb, int ldc, int nb1, int nb2,
                 long long sA1, long long sB1, long long sC1, long long sA2,
                 long long sB2, long long sC2) {
  dim3 g(N / BN, M / BM, nb1 * nb2), blk(256);
  if (transb) {
    if (outbf)
      k_gemm<true, true><<<g, blk, 0, st>>>(A, B, C, bias, M, N, K, lda, ldb,
                                            ldc, sA1, sB1, sC1, sA2, sB2, sC2, nb2);
    else
      k_gemm<true, false><<<g, blk, 0, st>>>(A, B, C, bias, M, N, K, lda, ldb,
                                             ldc, sA1, sB1, sC1, sA2, sB2, sC2, nb2);
  } else {
    if (outbf)
      k_gemm<false, true><<<g, blk, 0, st>>>(A, B, C, bias, M, N, K, lda, ldb,
                                             ldc, sA1, sB1, sC1, sA2, sB2, sC2, nb2);
    else
      k_gemm<false, false><<<g, blk, 0, st>>>(A, B, C, bias, M, N, K, lda, ldb,
                                              ldc, sA1, sB1, sC1, sA2, sB2, sC2, nb2);
  }
}

extern "C" void kernel_launch(void* const* d_in, const int* in_sizes, int n_in,
                              void* d_out, int out_size, void* d_ws,
                              size_t ws_size, hipStream_t stream) {
  (void)in_sizes; (void)n_in; (void)out_size; (void)ws_size;
  const float* x        = (const float*)d_in[0];
  const float* conv_w   = (const float*)d_in[1];
  const float* conv_b   = (const float*)d_in[2];
  const float* pos_emb  = (const float*)d_in[3];
  const float* latents  = (const float*)d_in[4];
  const float* ca_wq    = (const float*)d_in[5];
  const float* ca_wk    = (const float*)d_in[6];
  const float* ca_wv    = (const float*)d_in[7];
  const float* ca_wu    = (const float*)d_in[8];
  const float* ca_bu    = (const float*)d_in[9];
  const float* ca_ln_g  = (const float*)d_in[10];
  const float* ca_ln_b  = (const float*)d_in[11];
  const float* blk_wq   = (const float*)d_in[12];
  const float* blk_wk   = (const float*)d_in[13];
  const float* blk_wv   = (const float*)d_in[14];
  const float* blk_wu   = (const float*)d_in[15];
  const float* blk_bu   = (const float*)d_in[16];
  const float* blk_ln1g = (const float*)d_in[17];
  const float* blk_ln1b = (const float*)d_in[18];
  const float* blk_w1   = (const float*)d_in[19];
  const float* blk_b1   = (const float*)d_in[20];
  const float* blk_w2   = (const float*)d_in[21];
  const float* blk_b2   = (const float*)d_in[22];
  const float* blk_ln2g = (const float*)d_in[23];
  const float* blk_ln2b = (const float*)d_in[24];
  const float* cls_w    = (const float*)d_in[25];
  const float* cls_b    = (const float*)d_in[26];
  float* out = (float*)d_out;

  // ---- workspace bump allocator (256B aligned) ----
  size_t off = 0;
  char* base = (char*)d_ws;
  auto alloc_bf = [&](size_t elems) -> bf16_t* {
    bf16_t* p = (bf16_t*)(base + off);
    off = (off + elems * sizeof(bf16_t) + 255) & ~(size_t)255;
    return p;
  };
  auto alloc_f = [&](size_t elems) -> float* {
    float* p = (float*)(base + off);
    off = (off + elems * sizeof(float) + 255) & ~(size_t)255;
    return p;
  };

  bf16_t* lat_bf  = alloc_bf(512 * 256);
  bf16_t* cawq_bf = alloc_bf(524288);
  bf16_t* cawk_bf = alloc_bf(524288);
  bf16_t* cawv_bf = alloc_bf(524288);
  bf16_t* cawu_bf = alloc_bf(524288);
  bf16_t* bwq_bf  = alloc_bf(4 * 524288);
  bf16_t* bwk_bf  = alloc_bf(4 * 524288);
  bf16_t* bwv_bf  = alloc_bf(4 * 524288);
  bf16_t* bwu_bf  = alloc_bf(4 * 524288);
  bf16_t* bw1_bf  = alloc_bf(4 * 262144);
  bf16_t* bw2_bf  = alloc_bf(4 * 262144);
  bf16_t* tok_bf  = alloc_bf((size_t)4 * 4096 * 256);   // (B,N,E)
  bf16_t* q_bf    = alloc_bf((size_t)2048 * 2048);      // cross:512x2048 / self:2048x2048
  bf16_t* k_bf    = alloc_bf((size_t)4096 * 2048);      // cross per-b / self 2048x2048
  bf16_t* v_bf    = alloc_bf((size_t)4096 * 2048);
  bf16_t* sc_bf   = alloc_bf((size_t)8 * 512 * 4096);   // cross per-b scores (>= self 32*512*512)
  bf16_t* attn_bf = alloc_bf((size_t)4 * 512 * 2048);   // (B,L,HS)
  bf16_t* z_bf    = alloc_bf((size_t)2048 * 256);
  bf16_t* ff1_bf  = alloc_bf((size_t)2048 * 1024);
  float* proj_f   = alloc_f((size_t)2048 * 256);
  float* z_f      = alloc_f((size_t)2048 * 256);
  float* ff1_f    = alloc_f((size_t)2048 * 1024);
  float* zmean    = alloc_f(4 * 256);

  auto cvt = [&](const float* s, bf16_t* d, long long n) {
    k_cvt_bf16<<<(unsigned)((n + 255) / 256), 256, 0, stream>>>(s, d, n);
  };

  // ---- weight / latent conversions ----
  cvt(latents, lat_bf, 131072);
  cvt(ca_wq, cawq_bf, 524288);
  cvt(ca_wk, cawk_bf, 524288);
  cvt(ca_wv, cawv_bf, 524288);
  cvt(ca_wu, cawu_bf, 524288);
  cvt(blk_wq, bwq_bf, 4 * 524288);
  cvt(blk_wk, bwk_bf, 4 * 524288);
  cvt(blk_wv, bwv_bf, 4 * 524288);
  cvt(blk_wu, bwu_bf, 4 * 524288);
  cvt(blk_w1, bw1_bf, 4 * 262144);
  cvt(blk_w2, bw2_bf, 4 * 262144);

  // ---- patchify: 1x1 conv + positional embedding -> bf16 tokens ----
  k_conv_pos<<<(4 * 4096 * 256) / 256, 256, 0, stream>>>(x, conv_w, conv_b,
                                                         pos_emb, tok_bf);

  const float SCALE = 0.17677669529663687f;  // 1/sqrt(E/H) = 1/sqrt(32)

  // ---- cross attention: q from latents (shared across batch) ----
  gemm(stream, false, true, lat_bf, cawq_bf, q_bf, nullptr, 512, 2048, 256,
       256, 2048, 2048, 1, 1, 0, 0, 0, 0, 0, 0);

  for (int b = 0; b < 4; ++b) {
    const bf16_t* tb = tok_bf + (long long)b * 4096 * 256;
    // K,V projections for this image: (4096x256)@(256x2048)
    gemm(stream, false, true, tb, cawk_bf, k_bf, nullptr, 4096, 2048, 256,
         256, 2048, 2048, 1, 1, 0, 0, 0, 0, 0, 0);
    gemm(stream, false, true, tb, cawv_bf, v_bf, nullptr, 4096, 2048, 256,
         256, 2048, 2048, 1, 1, 0, 0, 0, 0, 0, 0);
    // scores[h] = Q_h (512x256) @ K_h^T (256x4096)
    gemm(stream, true, true, q_bf, k_bf, sc_bf, nullptr, 512, 4096, 256,
         2048, 2048, 4096, 8, 1,
         256, 256, (long long)512 * 4096, 0, 0, 0);
    k_softmax<16><<<8 * 512, 256, 0, stream>>>(sc_bf, SCALE);
    // attn[h] = P_h (512x4096) @ V_h (4096x256)  -> attn_bf[b, q, h*256+e]
    gemm(stream, false, true, sc_bf, v_bf,
         attn_bf + (long long)b * 512 * 2048, nullptr, 512, 256, 4096,
         4096, 2048, 2048, 8, 1,
         (long long)512 * 4096, 256, 256, 0, 0, 0);
  }
  // output projection + residual(latents broadcast) + LN
  gemm(stream, false, false, attn_bf, cawu_bf, proj_f, ca_bu, 2048, 256, 2048,
       2048, 256, 256, 1, 1, 0, 0, 0, 0, 0, 0);
  k_add_ln<<<2048, 256, 0, stream>>>(proj_f, latents, 512, ca_ln_g, ca_ln_b,
                                     z_f, z_bf);

  // ---- transformer blocks on latents (B*L = 2048 rows) ----
  for (int d = 0; d < 4; ++d) {
    const bf16_t* wq = bwq_bf + (long long)d * 524288;
    const bf16_t* wk = bwk_bf + (long long)d * 524288;
    const bf16_t* wv = bwv_bf + (long long)d * 524288;
    const bf16_t* wu = bwu_bf + (long long)d * 524288;
    gemm(stream, false, true, z_bf, wq, q_bf, nullptr, 2048, 2048, 256,
         256, 2048, 2048, 1, 1, 0, 0, 0, 0, 0, 0);
    gemm(stream, false, true, z_bf, wk, k_bf, nullptr, 2048, 2048, 256,
         256, 2048, 2048, 1, 1, 0, 0, 0, 0, 0, 0);
    gemm(stream, false, true, z_bf, wv, v_bf, nullptr, 2048, 2048, 256,
         256, 2048, 2048, 1, 1, 0, 0, 0, 0, 0, 0);
    // scores[b,h] = Q (512x256) @ K^T -> sc_bf[(b*8+h)*512*512]
    gemm(stream, true, true, q_bf, k_bf, sc_bf, nullptr, 512, 512, 256,
         2048, 2048, 512, 4, 8,
         (long long)512 * 2048, (long long)512 * 2048, (long long)8 * 512 * 512,
         256, 256, (long long)512 * 512);
    k_softmax<2><<<4 * 8 * 512, 256, 0, stream>>>(sc_bf, SCALE);
    // attn[b,h] = P (512x512) @ V_h (512x256)
    gemm(stream, false, true, sc_bf, v_bf, attn_bf, nullptr, 512, 256, 512,
         512, 2048, 2048, 4, 8,
         (long long)8 * 512 * 512, (long long)512 * 2048, (long long)512 * 2048,
         (long long)512 * 512, 256, 256);
    gemm(stream, false, false, attn_bf, wu, proj_f, blk_bu + d * 256,
         2048, 256, 2048, 2048, 256, 256, 1, 1, 0, 0, 0, 0, 0, 0);
    k_add_ln<<<2048, 256, 0, stream>>>(proj_f, z_f, 0, blk_ln1g + d * 256,
                                       blk_ln1b + d * 256, z_f, z_bf);
    // FFN
    gemm(stream, false, false, z_bf, bw1_bf + (long long)d * 262144, ff1_f,
         blk_b1 + d * 1024, 2048, 1024, 256, 256, 1024, 1024, 1, 1,
         0, 0, 0, 0, 0, 0);
    k_gelu_bf16<<<(2048 * 1024) / 256, 256, 0, stream>>>(ff1_f, ff1_bf,
                                                         (long long)2048 * 1024);
    gemm(stream, false, false, ff1_bf, bw2_bf + (long long)d * 262144, proj_f,
         blk_b2 + d * 256, 2048, 256, 1024, 1024, 256, 256, 1, 1,
         0, 0, 0, 0, 0, 0);
    k_add_ln<<<2048, 256, 0, stream>>>(proj_f, z_f, 0, blk_ln2g + d * 256,
                                       blk_ln2b + d * 256, z_f, z_bf);
  }

  // ---- mean over latents + classifier ----
  k_mean<<<4, 256, 0, stream>>>(z_f, zmean);
  k_cls<<<dim3(4, 4), 256, 0, stream>>>(zmean, cls_w, cls_b, out);
}
